// GNN_20366734917765
// MI455X (gfx1250) — matrix operand order repaired and checked by reading, not compiled, not run on hardware
//
#include <hip/hip_runtime.h>
#include <math.h>
#include <stdint.h>

// ---------------- problem constants ----------------
constexpr int BB   = 4;
constexpr int NP   = 2048;     // particles per batch
constexpr int DIM  = 3;
constexpr int Hd   = 128;      // hidden
constexpr int Kn   = 32;       // neighbours
constexpr int LL   = 4;        // layers
constexpr int OUTD = 6;
constexpr int BN   = BB * NP;  // 8192 rows total

typedef __bf16 bf16_t;
typedef bf16_t v16bf __attribute__((ext_vector_type(16)));
typedef float  v8f   __attribute__((ext_vector_type(8)));

#define FBIG 3.402823466e+38f

// ---------------- WMMA helpers ----------------
__device__ __forceinline__ v8f wmma_bf16(v16bf a, v16bf b, v8f c) {
    // D = A(16x32) * B(32x16) + C, f32 accumulate
    return __builtin_amdgcn_wmma_f32_16x16x32_bf16(
        false, a, false, b, (short)0, c, false, false);
}

// A fragment (16x32 bf16) from an LDS row-major matrix, row stride lda
// (elements). Layout: lane<16 holds row m0+lane, K {k0..k0+15};
// lane>=16 holds row m0+lane-16, K {k0+16..k0+31}.
__device__ __forceinline__ v16bf load_a_frag(const bf16_t* base, int lda,
                                             int m0, int k0, int lane) {
    int m  = m0 + (lane & 15);
    int kb = k0 + ((lane >> 4) << 4);
    const uint4* p = reinterpret_cast<const uint4*>(base + m * lda + kb);
    union { v16bf v; uint4 q[2]; } f;
    f.q[0] = p[0];     // K kb..kb+7
    f.q[1] = p[1];     // K kb+8..kb+15
    return f.v;
}

// B fragment (32x16 bf16) from pre-swizzled weights: per (k-chunk kc, n-tile nt)
// each lane owns 16 contiguous bf16 (32 bytes): column n = nt*16 + (lane&15),
// K = kc*32 + (lane>>4)*16 + e.
__device__ __forceinline__ v16bf load_b_frag(const bf16_t* wsw, int ntiles_n,
                                             int kc, int nt, int lane) {
    const uint4* p = reinterpret_cast<const uint4*>(
        wsw + (((size_t)(kc * ntiles_n + nt) * 32 + lane) << 4));
    union { v16bf v; uint4 q[2]; } f;
    f.q[0] = p[0];
    f.q[1] = p[1];
    return f.v;
}

__device__ __forceinline__ float silu_f(float x) {
    return x / (1.0f + __expf(-x));
}

// ---- CDNA5 async global->LDS copy (ASYNCcnt path, bypasses VGPRs) ----
// Per-lane: copies 16 bytes from gsrc to the LDS byte address given by the
// low 32 bits of the flat shared pointer (LDS aperture keeps offset in [31:0]).
__device__ __forceinline__ void async_copy_b128(void* lds_dst, const void* gsrc) {
    uint32_t lo = (uint32_t)(uintptr_t)lds_dst;
    asm volatile("global_load_async_to_lds_b128 %0, %1, off"
                 :: "v"(lo), "v"(gsrc) : "memory");
}
__device__ __forceinline__ void wait_async0() {
    asm volatile("s_wait_asynccnt 0" ::: "memory");
}

// ---------------- small prep kernels ----------------
__global__ void embed_kernel(const float* __restrict__ x,
                             const float* __restrict__ ew,
                             const float* __restrict__ eb,
                             float* __restrict__ h, bf16_t* __restrict__ hbf) {
    int i = blockIdx.x * blockDim.x + threadIdx.x;
    if (i >= BN * Hd) return;
    int row = i >> 7, c = i & 127;
    float r = eb[c];
    #pragma unroll
    for (int d = 0; d < DIM; ++d) r += x[row * DIM + d] * ew[d * Hd + c];
    h[i] = r;
    hbf[i] = (bf16_t)r;
}

// swizzle a (K x N) f32 weight (K mult of 32, N mult of 16) into B-fragment order
__global__ void swizzle_w(const float* __restrict__ src, bf16_t* __restrict__ dst,
                          int K, int Nn) {
    int i = blockIdx.x * blockDim.x + threadIdx.x;
    if (i >= K * Nn) return;
    int k = i / Nn, n = i % Nn;
    int kc = k >> 5, kin = k & 31;
    int lane = ((kin >> 4) << 4) + (n & 15);
    int e = kin & 15;
    int nt = n >> 4;
    size_t off = (((size_t)(kc * (Nn >> 4) + nt) * 32 + lane) << 4) + e;
    dst[off] = (bf16_t)src[i];
}

__global__ void f2bf_vec(const float* __restrict__ src, bf16_t* __restrict__ dst,
                         int n) {
    int i = blockIdx.x * blockDim.x + threadIdx.x;
    if (i < n) dst[i] = (bf16_t)src[i];
}

// ---------------- top-K neighbours ----------------
__global__ __launch_bounds__(256)
void topk_kernel(const float* __restrict__ x, int* __restrict__ nn_idx,
                 float* __restrict__ nn_d) {
    __shared__ float d2[NP];
    __shared__ float rv[256];
    __shared__ int   ri[256];
    int p = blockIdx.x;            // flat particle
    int b = p >> 11;               // batch
    int i = p & (NP - 1);
    int t = threadIdx.x;
    float xi0 = x[(size_t)p * 3 + 0];
    float xi1 = x[(size_t)p * 3 + 1];
    float xi2 = x[(size_t)p * 3 + 2];
    for (int j = t; j < NP; j += 256) {
        const float* xj = x + ((size_t)b * NP + j) * 3;
        float dx = xi0 - xj[0], dy = xi1 - xj[1], dz = xi2 - xj[2];
        d2[j] = (j == i) ? FBIG : (dx * dx + dy * dy + dz * dz);
    }
    __syncthreads();
    for (int k = 0; k < Kn; ++k) {
        float bv = FBIG; int bi = 0x7fffffff;
        for (int j = t; j < NP; j += 256) {
            float v = d2[j];
            if (v < bv) { bv = v; bi = j; }
        }
        rv[t] = bv; ri[t] = bi;
        __syncthreads();
        for (int s = 128; s > 0; s >>= 1) {
            if (t < s) {
                float ov = rv[t + s]; int oi = ri[t + s];
                if (ov < rv[t] || (ov == rv[t] && oi < ri[t])) {
                    rv[t] = ov; ri[t] = oi;
                }
            }
            __syncthreads();
        }
        if (t == 0) {
            nn_idx[(size_t)p * Kn + k] = ri[0];
            nn_d[(size_t)p * Kn + k]   = sqrtf(fmaxf(rv[0], 0.0f));
            d2[ri[0]] = FBIG;
        }
        __syncthreads();
    }
}

// ---------------- message + aggregate (WMMA GEMM1 + col-sum + matvec) -------
// One block (4 waves) per particle. A = [h_i | h_j] (32 rows x 256), +d column.
__global__ __launch_bounds__(128)
void msg_kernel(const bf16_t* __restrict__ hbf,
                const float* __restrict__ b1,       // msg_in_b[l]  (128)
                const float* __restrict__ W2,       // msg_out_w[l] f32 (128x128)
                const float* __restrict__ b2,       // msg_out_b[l] (128)
                const bf16_t* __restrict__ w1sw,    // swizzled 256x128 + d-col
                const int* __restrict__ nn_idx,
                const float* __restrict__ nn_d,
                bf16_t* __restrict__ aggbf) {
    __shared__ __align__(16) bf16_t As[Kn][2 * Hd];  // 32 x 256 bf16 = 16KB
    __shared__ float dS[Kn];
    __shared__ float sS[Hd];                         // column sums

    int p = blockIdx.x;
    int b = p >> 11;
    int t = threadIdx.x;
    int lane = t & 31, wv = t >> 5;

    // gathered neighbours (cols 128..255): async DMA straight into LDS.
    // 32 rows x 256B = 512 chunks of 16B; 128 lanes x 4 issues.
    #pragma unroll
    for (int it = 0; it < 4; ++it) {
        int e = t + it * 128;
        int r = e >> 4;                 // 16 chunks per row
        int c8 = (e & 15) << 3;         // bf16 element offset within row
        int j = nn_idx[(size_t)p * Kn + r];
        async_copy_b128(&As[r][Hd + c8],
                        hbf + ((size_t)b * NP + j) * Hd + c8);
    }
    // stage h_i broadcast (cols 0..127)
    {
        bf16_t hib = hbf[(size_t)p * Hd + t];
        #pragma unroll
        for (int r = 0; r < Kn; ++r) As[r][t] = hib;
    }
    if (t < Kn) dS[t] = nn_d[(size_t)p * Kn + t];
    sS[t] = 0.0f;
    wait_async0();
    __syncthreads();

    // GEMM1: (32 x 257) @ (257 x 128). Wave wv owns cols [wv*32, wv*32+32).
    v8f acc[2][2];
    int hi16 = lane >> 4;
    #pragma unroll
    for (int mt = 0; mt < 2; ++mt)
        #pragma unroll
        for (int nt = 0; nt < 2; ++nt) {
            float bias = b1[wv * 32 + nt * 16 + (lane & 15)];
            #pragma unroll
            for (int v = 0; v < 8; ++v) acc[mt][nt][v] = bias;
        }
    #pragma unroll
    for (int kc = 0; kc < 8; ++kc) {
        v16bf a0 = load_a_frag(&As[0][0], 2 * Hd, 0,  kc * 32, lane);
        v16bf a1 = load_a_frag(&As[0][0], 2 * Hd, 16, kc * 32, lane);
        v16bf bb0 = load_b_frag(w1sw, 8, kc, wv * 2 + 0, lane);
        v16bf bb1 = load_b_frag(w1sw, 8, kc, wv * 2 + 1, lane);
        acc[0][0] = wmma_bf16(a0, bb0, acc[0][0]);
        acc[0][1] = wmma_bf16(a0, bb1, acc[0][1]);
        acc[1][0] = wmma_bf16(a1, bb0, acc[1][0]);
        acc[1][1] = wmma_bf16(a1, bb1, acc[1][1]);
    }
    // rank-1 distance column (W1 row 256), then silu + column sum
    const bf16_t* wd = w1sw + 256 * Hd;
    #pragma unroll
    for (int mt = 0; mt < 2; ++mt)
        #pragma unroll
        for (int nt = 0; nt < 2; ++nt) {
            int col = wv * 32 + nt * 16 + (lane & 15);
            float wdc = (float)wd[col];
            float part = 0.0f;
            #pragma unroll
            for (int v = 0; v < 8; ++v) {
                int m = mt * 16 + hi16 * 8 + v;
                float xv = acc[mt][nt][v] + dS[m] * wdc;
                part += silu_f(xv);
            }
            atomicAdd(&sS[col], part);
        }
    __syncthreads();

    // matvec: agg[n] = sum_k sS[k]*W2[k][n] + Kn*b2[n]   (agg stored bf16:
    // it only ever feeds a bf16 WMMA A-operand downstream)
    float r = (float)Kn * b2[t];
    for (int k = 0; k < Hd; ++k) r += sS[k] * W2[k * Hd + t];
    aggbf[(size_t)p * Hd + t] = (bf16_t)r;
}

// ---------------- update MLP (two chained WMMA GEMMs) ----------------
__global__ __launch_bounds__(128)
void upd_kernel(float* __restrict__ h, bf16_t* __restrict__ hbf,
                const bf16_t* __restrict__ aggbf,
                const bf16_t* __restrict__ wu1sw,  // 256x128 swizzled
                const float* __restrict__ bu1,
                const bf16_t* __restrict__ wu2sw,  // 128x128 swizzled
                const float* __restrict__ bu2) {
    __shared__ __align__(16) bf16_t As[32][2 * Hd];  // 16KB
    __shared__ __align__(16) bf16_t T1[32][Hd];      // 8KB
    int row0 = blockIdx.x * 32;
    int t = threadIdx.x;
    int lane = t & 31, wv = t >> 5;
    int hi16 = lane >> 4;

    // stage [h | agg] rows via async DMA to LDS (2 x 8KB)
    #pragma unroll
    for (int it = 0; it < 4; ++it) {
        int e = t + it * 128;
        int r = e >> 4;
        int c8 = (e & 15) << 3;
        size_t go = (size_t)(row0 + r) * Hd + c8;
        async_copy_b128(&As[r][c8],      hbf   + go);
        async_copy_b128(&As[r][Hd + c8], aggbf + go);
    }
    wait_async0();
    __syncthreads();

    // GEMM1: (32 x 256) @ (256 x 128), silu -> T1 (bf16)
    {
        v8f acc[2][2];
        #pragma unroll
        for (int mt = 0; mt < 2; ++mt)
            #pragma unroll
            for (int nt = 0; nt < 2; ++nt) {
                float bias = bu1[wv * 32 + nt * 16 + (lane & 15)];
                #pragma unroll
                for (int v = 0; v < 8; ++v) acc[mt][nt][v] = bias;
            }
        #pragma unroll
        for (int kc = 0; kc < 8; ++kc) {
            v16bf a0 = load_a_frag(&As[0][0], 2 * Hd, 0,  kc * 32, lane);
            v16bf a1 = load_a_frag(&As[0][0], 2 * Hd, 16, kc * 32, lane);
            v16bf bb0 = load_b_frag(wu1sw, 8, kc, wv * 2 + 0, lane);
            v16bf bb1 = load_b_frag(wu1sw, 8, kc, wv * 2 + 1, lane);
            acc[0][0] = wmma_bf16(a0, bb0, acc[0][0]);
            acc[0][1] = wmma_bf16(a0, bb1, acc[0][1]);
            acc[1][0] = wmma_bf16(a1, bb0, acc[1][0]);
            acc[1][1] = wmma_bf16(a1, bb1, acc[1][1]);
        }
        #pragma unroll
        for (int mt = 0; mt < 2; ++mt)
            #pragma unroll
            for (int nt = 0; nt < 2; ++nt) {
                int col = wv * 32 + nt * 16 + (lane & 15);
                #pragma unroll
                for (int v = 0; v < 8; ++v) {
                    int m = mt * 16 + hi16 * 8 + v;
                    T1[m][col] = (bf16_t)silu_f(acc[mt][nt][v]);
                }
            }
    }
    __syncthreads();

    // GEMM2: (32 x 128) @ (128 x 128), residual add into h (+ refresh hbf)
    {
        v8f acc[2][2] = {};
        #pragma unroll
        for (int kc = 0; kc < 4; ++kc) {
            v16bf a0 = load_a_frag(&T1[0][0], Hd, 0,  kc * 32, lane);
            v16bf a1 = load_a_frag(&T1[0][0], Hd, 16, kc * 32, lane);
            v16bf bb0 = load_b_frag(wu2sw, 8, kc, wv * 2 + 0, lane);
            v16bf bb1 = load_b_frag(wu2sw, 8, kc, wv * 2 + 1, lane);
            acc[0][0] = wmma_bf16(a0, bb0, acc[0][0]);
            acc[0][1] = wmma_bf16(a0, bb1, acc[0][1]);
            acc[1][0] = wmma_bf16(a1, bb0, acc[1][0]);
            acc[1][1] = wmma_bf16(a1, bb1, acc[1][1]);
        }
        #pragma unroll
        for (int mt = 0; mt < 2; ++mt)
            #pragma unroll
            for (int nt = 0; nt < 2; ++nt) {
                int col = wv * 32 + nt * 16 + (lane & 15);
                float bias = bu2[col];
                #pragma unroll
                for (int v = 0; v < 8; ++v) {
                    int m = mt * 16 + hi16 * 8 + v;
                    size_t o = (size_t)(row0 + m) * Hd + col;
                    float nh = h[o] + acc[mt][nt][v] + bias;
                    h[o] = nh;
                    hbf[o] = (bf16_t)nh;
                }
            }
    }
}

// ---------------- output head ----------------
__global__ void out_kernel(const float* __restrict__ h,
                           const float* __restrict__ ow,
                           const float* __restrict__ ob,
                           float* __restrict__ out) {
    int i = blockIdx.x * blockDim.x + threadIdx.x;
    if (i >= BN * OUTD) return;
    int row = i / OUTD, o = i % OUTD;
    float r = ob[o];
    for (int k = 0; k < Hd; ++k) r += h[(size_t)row * Hd + k] * ow[k * OUTD + o];
    out[i] = r;
}

// ---------------- launcher ----------------
extern "C" void kernel_launch(void* const* d_in, const int* in_sizes, int n_in,
                              void* d_out, int out_size, void* d_ws, size_t ws_size,
                              hipStream_t stream) {
    (void)in_sizes; (void)n_in; (void)out_size; (void)ws_size;
    const float* x        = (const float*)d_in[0];
    const float* embed_w  = (const float*)d_in[1];
    const float* embed_b  = (const float*)d_in[2];
    const float* msg_in_w = (const float*)d_in[3];   // (L,257,128)
    const float* msg_in_b = (const float*)d_in[4];   // (L,128)
    const float* msg_out_w= (const float*)d_in[5];   // (L,128,128)
    const float* msg_out_b= (const float*)d_in[6];   // (L,128)
    const float* upd_in_w = (const float*)d_in[7];   // (L,256,128)
    const float* upd_in_b = (const float*)d_in[8];   // (L,128)
    const float* upd_out_w= (const float*)d_in[9];   // (L,128,128)
    const float* upd_out_b= (const float*)d_in[10];  // (L,128)
    const float* out_w    = (const float*)d_in[11];  // (128,6)
    const float* out_b    = (const float*)d_in[12];  // (6)

    char* ws = (char*)d_ws;
    size_t off = 0;
    auto alloc = [&](size_t bytes) -> void* {
        off = (off + 255) & ~(size_t)255;
        void* p = ws + off;
        off += bytes;
        return p;
    };
    float*  h     = (float*)alloc((size_t)BN * Hd * 4);
    bf16_t* hbf   = (bf16_t*)alloc((size_t)BN * Hd * 2);
    bf16_t* aggbf = (bf16_t*)alloc((size_t)BN * Hd * 2);
    int*    nnidx = (int*)alloc((size_t)BN * Kn * 4);
    float*  nnd   = (float*)alloc((size_t)BN * Kn * 4);
    const size_t W1L = 256 * Hd + Hd;          // swizzled + d-column
    bf16_t* w1sw  = (bf16_t*)alloc((size_t)LL * W1L * 2);
    bf16_t* wu1sw = (bf16_t*)alloc((size_t)LL * 256 * Hd * 2);
    bf16_t* wu2sw = (bf16_t*)alloc((size_t)LL * 128 * Hd * 2);

    embed_kernel<<<(BN * Hd + 255) / 256, 256, 0, stream>>>(x, embed_w, embed_b, h, hbf);
    topk_kernel<<<BN, 256, 0, stream>>>(x, nnidx, nnd);

    for (int l = 0; l < LL; ++l) {
        swizzle_w<<<(256 * Hd + 255) / 256, 256, 0, stream>>>(
            msg_in_w + (size_t)l * 257 * Hd, w1sw + (size_t)l * W1L, 256, Hd);
        f2bf_vec<<<1, Hd, 0, stream>>>(
            msg_in_w + (size_t)l * 257 * Hd + 256 * Hd,
            w1sw + (size_t)l * W1L + 256 * Hd, Hd);
        swizzle_w<<<(256 * Hd + 255) / 256, 256, 0, stream>>>(
            upd_in_w + (size_t)l * 256 * Hd, wu1sw + (size_t)l * 256 * Hd, 256, Hd);
        swizzle_w<<<(128 * Hd + 255) / 256, 256, 0, stream>>>(
            upd_out_w + (size_t)l * 128 * Hd, wu2sw + (size_t)l * 128 * Hd, 128, Hd);
    }

    for (int l = 0; l < LL; ++l) {
        msg_kernel<<<BN, 128, 0, stream>>>(
            hbf,
            msg_in_b + (size_t)l * Hd,
            msg_out_w + (size_t)l * Hd * Hd,
            msg_out_b + (size_t)l * Hd,
            w1sw + (size_t)l * W1L,
            nnidx, nnd, aggbf);
        upd_kernel<<<BN / 32, 128, 0, stream>>>(
            h, hbf, aggbf,
            wu1sw + (size_t)l * 256 * Hd,
            upd_in_b + (size_t)l * Hd,
            wu2sw + (size_t)l * 128 * Hd,
            upd_out_b + (size_t)l * Hd);
    }

    out_kernel<<<(BN * OUTD + 255) / 256, 256, 0, stream>>>(h, out_w, out_b, (float*)d_out);
}
